// NumberMapping_49452253446290
// MI455X (gfx1250) — 8-bit matrix operand order repaired and checked by reading, compiled, not run on hardware
//
#include <hip/hip_runtime.h>
#include <cstddef>

typedef int v8i __attribute__((ext_vector_type(8)));

#define BATCH   32768
#define HID     2048
#define H2      1024
#define NDIM    16
#define NPROTO  101

__device__ __forceinline__ v8i load_a_frag(const signed char* ap) {
    int2 t0 = *(const int2*)(ap);
    int2 t1 = *(const int2*)(ap + 16);
    int2 t2 = *(const int2*)(ap + 32);
    int2 t3 = *(const int2*)(ap + 48);
    v8i a;
    a[0] = t0.x; a[1] = t0.y; a[2] = t1.x; a[3] = t1.y;
    a[4] = t2.x; a[5] = t2.y; a[6] = t3.x; a[7] = t3.y;
    return a;
}

__device__ __forceinline__ v8i load_b_frag_lds(const unsigned char* bp) {
    int2 t0 = *(const int2*)(bp);
    int2 t1 = *(const int2*)(bp + 16);
    int2 t2 = *(const int2*)(bp + 32);
    int2 t3 = *(const int2*)(bp + 48);
    v8i b;
    b[0] = t0.x; b[1] = t0.y; b[2] = t1.x; b[3] = t1.y;
    b[4] = t2.x; b[5] = t2.y; b[6] = t3.x; b[7] = t3.y;
    return b;
}

__global__ __launch_bounds__(256) void abs_sum_kernel(const float* __restrict__ x,
                                                      long long n,
                                                      float* __restrict__ out) {
    __shared__ float red[8];
    float s = 0.0f;
    for (long long i = (long long)blockIdx.x * 256 + threadIdx.x; i < n;
         i += (long long)gridDim.x * 256)
        s += fabsf(x[i]);
    for (int o = 16; o > 0; o >>= 1) s += __shfl_xor(s, o, 32);
    if ((threadIdx.x & 31) == 0) red[threadIdx.x >> 5] = s;
    __syncthreads();
    if (threadIdx.x == 0) {
        float t = 0.0f;
        for (int i = 0; i < 8; ++i) t += red[i];
        atomicAdd(out, t);
    }
}

__global__ __launch_bounds__(256) void quant_weights_kernel(const float* __restrict__ w,
                                                            signed char* __restrict__ wq,
                                                            const float* __restrict__ sum,
                                                            float inv_count, int n) {
    const float beta = *sum * inv_count;
    const float inv  = 1.0f / (beta + 1e-8f);
    for (int i = blockIdx.x * 256 + threadIdx.x; i < n; i += gridDim.x * 256) {
        float v = rintf(w[i] * inv);
        v = fminf(fmaxf(v, -1.0f), 1.0f);
        wq[i] = (signed char)v;
    }
}

__global__ __launch_bounds__(256) void quant_rows_kernel(const float* __restrict__ x,
                                                         signed char* __restrict__ q,
                                                         float* __restrict__ s_out,
                                                         int rowlen) {
    __shared__ float red[8];
    const int row = blockIdx.x;
    const float* xr = x + (size_t)row * rowlen;
    const int per  = rowlen >> 8;
    const int base = threadIdx.x * per;

    float m = 0.0f;
    for (int j = 0; j < per; ++j) m = fmaxf(m, fabsf(xr[base + j]));
    for (int o = 16; o > 0; o >>= 1) m = fmaxf(m, __shfl_xor(m, o, 32));
    if ((threadIdx.x & 31) == 0) red[threadIdx.x >> 5] = m;
    __syncthreads();
    float bm = red[0];
#pragma unroll
    for (int i = 1; i < 8; ++i) bm = fmaxf(bm, red[i]);

    const float s = 127.0f / fmaxf(bm, 1e-5f);
    signed char* qr = q + (size_t)row * rowlen + base;
    for (int j = 0; j < per; ++j) {
        float v = rintf(xr[base + j] * s);
        v = fminf(fmaxf(v, -128.0f), 127.0f);
        qr[j] = (signed char)v;
    }
    if (threadIdx.x == 0) s_out[row] = s;
}

__global__ __launch_bounds__(256) void gemm1_gelu_kernel(const signed char* __restrict__ xq,
                                                         const signed char* __restrict__ w1q,
                                                         const float* __restrict__ s1,
                                                         const float* __restrict__ b1,
                                                         const float* __restrict__ betas,
                                                         float* __restrict__ h) {
    __shared__ alignas(16) unsigned char bs[2][64 * 80];

    const int lane   = threadIdx.x & 31;
    const int wave   = threadIdx.x >> 5;
    const int m_base = blockIdx.x * 256 + wave * 32;
    const int n0     = blockIdx.y * 64;

    const int khalf  = (lane >> 4) * 8;
    const int nlane  = lane & 15;
    const int kq     = (lane >> 4) << 3;
    const signed char* arow0 = xq + (size_t)(m_base + (lane & 15)) * HID;
    const signed char* arow1 = arow0 + (size_t)16 * HID;

    const int snl = threadIdx.x >> 2;
    const int skl = (threadIdx.x & 3) << 4;
    const signed char* wsrc = w1q + (size_t)(n0 + snl) * HID + skl;

    v8i acc[2][4] = {};

    *(int4*)(&bs[0][snl * 80 + skl]) = *(const int4*)(wsrc);

    int cur = 0;
    for (int k0 = 0; k0 < HID; k0 += 64) {
        __syncthreads();
        if (k0 + 64 < HID)
            *(int4*)(&bs[cur ^ 1][snl * 80 + skl]) = *(const int4*)(wsrc + k0 + 64);

        const signed char* ap0 = arow0 + k0 + khalf;
        const signed char* ap1 = arow1 + k0 + khalf;
        __builtin_prefetch(ap0 + 64, 0, 1);
        __builtin_prefetch(ap1 + 64, 0, 1);
        v8i a0 = load_a_frag(ap0);
        v8i a1 = load_a_frag(ap1);

        const unsigned char* brow = &bs[cur][nlane * 80 + kq];
        v8i bf[4];
#pragma unroll
        for (int nt = 0; nt < 4; ++nt)
            bf[nt] = load_b_frag_lds(brow + nt * 16 * 80);

#pragma unroll
        for (int nt = 0; nt < 4; ++nt) {
            acc[0][nt] = __builtin_amdgcn_wmma_i32_16x16x64_iu8(
                true, a0, true, bf[nt], acc[0][nt], false, false);
            acc[1][nt] = __builtin_amdgcn_wmma_i32_16x16x64_iu8(
                true, a1, true, bf[nt], acc[1][nt], false, false);
        }
        cur ^= 1;
    }

    const float beta1 = betas[0] * (1.0f / (float)(H2 * HID));
    const int   mhalf = (lane >> 4) * 8;
#pragma unroll
    for (int mt = 0; mt < 2; ++mt) {
        const int mb = m_base + mt * 16 + mhalf;
        float rs[8];
#pragma unroll
        for (int v = 0; v < 8; ++v) rs[v] = beta1 / s1[mb + v];
#pragma unroll
        for (int nt = 0; nt < 4; ++nt) {
            const int n = n0 + nt * 16 + nlane;
            const float bias = b1[n];
#pragma unroll
            for (int v = 0; v < 8; ++v) {
                float val = (float)acc[mt][nt][v] * rs[v] + bias;
                float g = 0.5f * val * (1.0f + erff(val * 0.70710678118654752f));
                h[(size_t)(mb + v) * H2 + n] = g;
            }
        }
    }
}

__global__ __launch_bounds__(256) void gemm2_kernel(const signed char* __restrict__ hq,
                                                    const signed char* __restrict__ w2q,
                                                    const float* __restrict__ s2,
                                                    const float* __restrict__ b2,
                                                    const float* __restrict__ betas,
                                                    float* __restrict__ out) {
    __shared__ alignas(16) unsigned char bs[16 * 1040];
    {
        const int row = threadIdx.x >> 4;
        const int ch  = (threadIdx.x & 15) << 6;
#pragma unroll
        for (int j = 0; j < 4; ++j)
            *(int4*)(&bs[row * 1040 + ch + j * 16]) =
                *(const int4*)(w2q + row * H2 + ch + j * 16);
    }
    __syncthreads();

    const int lane   = threadIdx.x & 31;
    const int wave   = threadIdx.x >> 5;
    const int m_base = blockIdx.x * 256 + wave * 32;
    const int khalf  = (lane >> 4) * 8;
    const int nlane  = lane & 15;
    const int kq     = (lane >> 4) << 3;
    const signed char* arow0 = hq + (size_t)(m_base + (lane & 15)) * H2;
    const signed char* arow1 = arow0 + (size_t)16 * H2;

    v8i acc[2] = {};
    for (int k0 = 0; k0 < H2; k0 += 64) {
        v8i a0 = load_a_frag(arow0 + k0 + khalf);
        v8i a1 = load_a_frag(arow1 + k0 + khalf);
        v8i b  = load_b_frag_lds(&bs[nlane * 1040 + k0 + kq]);
        acc[0] = __builtin_amdgcn_wmma_i32_16x16x64_iu8(true, a0, true, b,
                                                        acc[0], false, false);
        acc[1] = __builtin_amdgcn_wmma_i32_16x16x64_iu8(true, a1, true, b,
                                                        acc[1], false, false);
    }

    const float beta2 = betas[1] * (1.0f / (float)(NDIM * H2));
    const int   mhalf = (lane >> 4) * 8;
    const float bias  = b2[nlane];
#pragma unroll
    for (int mt = 0; mt < 2; ++mt) {
        const int mb = m_base + mt * 16 + mhalf;
#pragma unroll
        for (int v = 0; v < 8; ++v) {
            const int m = mb + v;
            out[(size_t)m * NDIM + nlane] = (float)acc[mt][v] * beta2 / s2[m] + bias;
        }
    }
}

__global__ __launch_bounds__(256) void argmax_kernel(const float* __restrict__ rep,
                                                     const float* __restrict__ protos,
                                                     float* __restrict__ idx_out,
                                                     int batch) {
    __shared__ float pn[NPROTO * NDIM];
    for (int p = threadIdx.x; p < NPROTO; p += 256) {
        float v[NDIM]; float nn = 0.0f;
#pragma unroll
        for (int j = 0; j < NDIM; ++j) { v[j] = protos[p * NDIM + j]; nn += v[j] * v[j]; }
        const float inv = 1.0f / fmaxf(sqrtf(nn), 1e-8f);
#pragma unroll
        for (int j = 0; j < NDIM; ++j) pn[p * NDIM + j] = v[j] * inv;
    }
    __syncthreads();

    const int r = blockIdx.x * 256 + threadIdx.x;
    if (r >= batch) return;

    float a[NDIM]; float nn = 0.0f;
    const float4* rp = (const float4*)(rep + (size_t)r * NDIM);
#pragma unroll
    for (int j4 = 0; j4 < 4; ++j4) {
        float4 v = rp[j4];
        a[4 * j4 + 0] = v.x; a[4 * j4 + 1] = v.y;
        a[4 * j4 + 2] = v.z; a[4 * j4 + 3] = v.w;
    }
#pragma unroll
    for (int j = 0; j < NDIM; ++j) nn += a[j] * a[j];
    const float inv = 1.0f / fmaxf(sqrtf(nn), 1e-8f);

    float best = -3.4e38f; int bi = 0;
    for (int p = 0; p < NPROTO; ++p) {
        float d = 0.0f;
#pragma unroll
        for (int j = 0; j < NDIM; ++j) d += a[j] * pn[p * NDIM + j];
        d *= inv;
        if (d > best) { best = d; bi = p; }
    }
    idx_out[r] = (float)bi;
}

extern "C" void kernel_launch(void* const* d_in, const int* in_sizes, int n_in,
                              void* d_out, int out_size, void* d_ws, size_t ws_size,
                              hipStream_t stream) {
    (void)in_sizes; (void)n_in; (void)out_size; (void)ws_size;
    const float* x      = (const float*)d_in[0];
    const float* W1     = (const float*)d_in[1];
    const float* b1     = (const float*)d_in[2];
    const float* W2     = (const float*)d_in[3];
    const float* b2     = (const float*)d_in[4];
    const float* protos = (const float*)d_in[5];
    float* out = (float*)d_out;

    char* ws = (char*)d_ws;
    float*       betas = (float*)ws;
    float*       s1    = (float*)(ws + 256);
    float*       s2    = (float*)(ws + 256 + 131072);
    signed char* w1q   = (signed char*)(ws + 262400);
    signed char* w2q   = (signed char*)(ws + 2359552);
    signed char* xq    = (signed char*)(ws + 2375936);
    signed char* hq    = (signed char*)(ws + 69484800);
    float*       h     = (float*)(ws + 103039232);

    hipMemsetAsync(betas, 0, 2 * sizeof(float), stream);

    abs_sum_kernel<<<512, 256, 0, stream>>>(W1, (long long)H2 * HID, betas + 0);
    abs_sum_kernel<<<64, 256, 0, stream>>>(W2, (long long)NDIM * H2, betas + 1);

    quant_weights_kernel<<<2048, 256, 0, stream>>>(W1, w1q, betas + 0,
                                                   1.0f / (float)(H2 * HID), H2 * HID);
    quant_weights_kernel<<<64, 256, 0, stream>>>(W2, w2q, betas + 1,
                                                 1.0f / (float)(NDIM * H2), NDIM * H2);

    quant_rows_kernel<<<BATCH, 256, 0, stream>>>(x, xq, s1, HID);

    gemm1_gelu_kernel<<<dim3(BATCH / 256, H2 / 64), 256, 0, stream>>>(
        xq, w1q, s1, b1, betas, h);

    quant_rows_kernel<<<BATCH, 256, 0, stream>>>(h, hq, s2, H2);

    gemm2_kernel<<<BATCH / 256, 256, 0, stream>>>(hq, w2q, s2, b2, betas, out);

    argmax_kernel<<<BATCH / 256, 256, 0, stream>>>(out, protos,
                                                   out + (size_t)BATCH * NDIM, BATCH);
}
